// CrossPolarAttention_4380866642491
// MI455X (gfx1250) — matrix operand order repaired and checked by reading, not compiled, not run on hardware
//
#include <hip/hip_runtime.h>
#include <math.h>

typedef __attribute__((ext_vector_type(16))) _Float16 v16h;
typedef __attribute__((ext_vector_type(4)))  _Float16 v4h;
typedef __attribute__((ext_vector_type(8)))  float    v8f;
typedef __attribute__((ext_vector_type(4)))  float    v4f;
typedef __attribute__((ext_vector_type(4)))  int      v4i_t;

#define DIMC  64
#define NHEAD 8
#define NPOL  4
#define HH    128
#define WW    128
#define HWSZ  (HH*WW)
#define NB    4
#define EPSN  1e-12f

// ---------------------------------------------------------------------------
// gfx1250 async global<->LDS copies (ASYNCcnt-tracked), with fallbacks.
// Builtins take generic pointers to 16B int vectors (probe-verified).
// ---------------------------------------------------------------------------
#if __has_builtin(__builtin_amdgcn_global_load_async_to_lds_b128) && \
    __has_builtin(__builtin_amdgcn_global_store_async_from_lds_b128)
#define HAVE_ASYNC 1
#else
#define HAVE_ASYNC 0
#endif

#if __has_builtin(__builtin_amdgcn_s_wait_asynccnt)
#define WAIT_ASYNC() __builtin_amdgcn_s_wait_asynccnt(0)
#else
#define WAIT_ASYNC() asm volatile("s_wait_asynccnt 0x0" ::: "memory")
#endif

#if HAVE_ASYNC
__device__ __forceinline__ void async_g2l_b128(const void* g, void* l) {
  __builtin_amdgcn_global_load_async_to_lds_b128(
      (v4i_t*)g, (v4i_t*)l, 0, 0);
}
__device__ __forceinline__ void async_l2g_b128(void* g, const void* l) {
  __builtin_amdgcn_global_store_async_from_lds_b128(
      (v4i_t*)g, (v4i_t*)l, 0, 0);
}
#endif

// ---------------------------------------------------------------------------
// Kernel 0: fold pointwise 1x1 conv + channel-mean into a single 64-vector.
// mean_o( pw_w[o,:] @ g + pw_b[o] ) = w_avg . g + b_avg
// ---------------------------------------------------------------------------
__global__ void prep_kernel(const float* __restrict__ pw_w,
                            const float* __restrict__ pw_b,
                            float* __restrict__ ws) {
  int i = threadIdx.x;
  if (i < DIMC) {
    float s = 0.f;
    for (int o = 0; o < DIMC; ++o) s += pw_w[o * DIMC + i];
    ws[i] = s * (1.0f / DIMC);
  }
  if (i == DIMC) {
    float s = 0.f;
    for (int o = 0; o < DIMC; ++o) s += pw_b[o];
    ws[DIMC] = s * (1.0f / DIMC);
  }
}

// ---------------------------------------------------------------------------
// Kernel 1: mask branch. Fused depthwise 3x3 (SAME, zero pad) + exact GELU
// + dot with w_avg. Output mb laid out [b][h][w][p].
// ---------------------------------------------------------------------------
__global__ __launch_bounds__(256)
void maskbias_kernel(const float* __restrict__ mask,
                     const float* __restrict__ dw_w,
                     const float* __restrict__ dw_b,
                     const float* __restrict__ ws,
                     float* __restrict__ mb) {
  int id = blockIdx.x * blockDim.x + threadIdx.x;   // over (n, h, w)
  int wp = id % WW;
  int t  = id / WW;
  int hp = t % HH;
  int n  = t / HH;                                  // n = b*NPOL + p

  const float* base  = mask + (size_t)n * DIMC * HWSZ;
  float b_avg = ws[DIMC];
  float dot = 0.f;
  for (int c = 0; c < DIMC; ++c) {
    const float* pl = base + (size_t)c * HWSZ;
    if (c + 1 < DIMC) {
      // gfx1250: global_prefetch_b8 — stream the next channel plane
      __builtin_prefetch(pl + HWSZ + hp * WW + wp, 0, 0);
    }
    float acc = dw_b[c];
#pragma unroll
    for (int dy = -1; dy <= 1; ++dy) {
      int hy = hp + dy;
      if (hy < 0 || hy >= HH) continue;
#pragma unroll
      for (int dx = -1; dx <= 1; ++dx) {
        int wx = wp + dx;
        if (wx < 0 || wx >= WW) continue;
        acc += pl[hy * WW + wx] * dw_w[c * 9 + (dy + 1) * 3 + (dx + 1)];
      }
    }
    float g = 0.5f * acc * (1.0f + erff(acc * 0.70710678118654752f));
    dot += ws[c] * g;
  }
  int b = n / NPOL, p = n % NPOL;
  mb[(((size_t)b * HH + hp) * WW + wp) * NPOL + p] = dot + b_avg;
}

// ---------------------------------------------------------------------------
// LDS swizzles matching gfx1250 WMMA f16 VGPR layouts (ISA 7.12.2).
// ---------------------------------------------------------------------------
__device__ __forceinline__ int a_swz(int r, int K, int* hh) {
  if (K < 8)       { *hh = K;      return r; }
  else if (K < 16) { *hh = K - 8;  return 16 + r; }
  else if (K < 24) { *hh = K - 8;  return r; }
  else             { *hh = K - 16; return 16 + r; }
}

// ---------------------------------------------------------------------------
// Kernel 2: per-pixel cross-polar attention.
// One 128-thread block = 16 consecutive w pixels of one (b, h) row.
// 4 waves; wave m owns M-tile m (rows = 4 pixels x 4 pols = 16).
// LDS map (64 KB, phase-aliased):
//   [0,   8K)  ldsA   : x tile (A layout), later attn-out (A layout)
//   [8K, 40K)  ldsB   : wq,wk,wv,wproj in B layout (f16)
//   [40K,56K)  rawX   : async-DMA'd raw f32 x tile [pol*64+ch][pix]
//   [40K,64K)  ldsQKV : q,k,v f16 [3][pix][pol][ch]
//   [40K,56K)  ldsOut : f32 output staging [pol*64+ch][pix] (async store src)
// ---------------------------------------------------------------------------
__global__ __launch_bounds__(128)
void attention_kernel(const float* __restrict__ x_in,
                      const float* __restrict__ wq,
                      const float* __restrict__ wk,
                      const float* __restrict__ wv,
                      const float* __restrict__ w_proj,
                      const float* __restrict__ b_proj,
                      const float* __restrict__ rescale,
                      const float* __restrict__ mb,
                      float* __restrict__ out) {
  __shared__ __attribute__((aligned(32))) unsigned char smem[65536];
  _Float16* ldsA   = (_Float16*)(smem);
  _Float16* ldsB   = (_Float16*)(smem + 8192);
  float*    rawX   = (float*)(smem + 40960);
  _Float16* ldsQKV = (_Float16*)(smem + 40960);
  float*    ldsOut = (float*)(smem + 40960);

  const int tid  = threadIdx.x;
  const int lane = tid & 31;
  const int wave = tid >> 5;          // == M-tile

  const int wt = blockIdx.x & 7;          // W / 16 = 8 tiles
  const int h  = (blockIdx.x >> 3) & 127;
  const int b  = blockIdx.x >> 10;
  const int w0 = wt * 16;

#if HAVE_ASYNC
  // ---- Phase 1a: async DMA of the raw f32 x tile into LDS ------------------
  {
#pragma unroll
    for (int it = 0; it < 8; ++it) {
      int chunk = it * 128 + tid;       // 1024 chunks of 16B
      int plane = chunk >> 2;
      int quad  = chunk & 3;
      int pol = plane >> 6;
      int ch  = plane & 63;
      const float* g = x_in + ((size_t)(b * NPOL + pol) * DIMC + ch) * HWSZ
                            + h * WW + w0 + quad * 4;
      async_g2l_b128(g, rawX + chunk * 4);
    }
  }
#endif

  // ---- Phase 1b: weights -> B layout (overlaps the x DMA) ------------------
  {
#pragma unroll
    for (int m = 0; m < 4; ++m) {
      const float* Wm = (m == 0) ? wq : (m == 1) ? wk : (m == 2) ? wv : w_proj;
#pragma unroll
      for (int it = 0; it < 8; ++it) {
        int idx = it * 512 + tid * 4;   // coalesced over [d][c]
        v4f w4 = *(const v4f*)(Wm + idx);
        int d = idx >> 6, c0 = idx & 63;   // B[K=c][N=d] = W[d][c]
        int nt = d >> 4, n = d & 15;
        int ks = c0 >> 5, K0 = c0 & 31;
        int ln = n + ((K0 & 8) << 1);
        int hh = (K0 & 7) | ((K0 & 16) >> 1);
        v4h h4;
#pragma unroll
        for (int e = 0; e < 4; ++e) h4[e] = (_Float16)w4[e];
        *(v4h*)(ldsB + ((((m * 4 + nt) * 2 + ks) * 32 + ln) << 4) + hh) = h4;
      }
    }
  }

#if HAVE_ASYNC
  WAIT_ASYNC();
  __syncthreads();
  // ---- Phase 1c: swizzle raw x (LDS) -> WMMA A layout, packed v4h ----------
  {
#pragma unroll
    for (int it = 0; it < 8; ++it) {
      int task = it * 128 + tid;        // 1024 tasks of 4 channels
      int pix = task & 15;
      int pol = (task >> 4) & 3;
      int cg  = task >> 6;
      int ch0 = cg * 4;
      v4h h4;
#pragma unroll
      for (int e = 0; e < 4; ++e)
        h4[e] = (_Float16)rawX[((pol * DIMC + ch0 + e) << 4) + pix];
      int mt = pix >> 2;
      int r  = ((pix & 3) << 2) | pol;
      int ks = ch0 >> 5;
      int hh;
      int ln = a_swz(r, ch0 & 31, &hh);
      *(v4h*)(ldsA + (((mt * 2 + ks) * 32 + ln) << 4) + hh) = h4;
    }
  }
  __syncthreads();
#else
  {
    const int hw  = tid >> 4;
    const int pix = tid & 15;
    float xv[32];
#pragma unroll
    for (int it = 0; it < 32; ++it) {
      int plane = it * 8 + hw;
      int pol = plane >> 6;
      int ch  = plane & 63;
      xv[it] = x_in[((size_t)(b * NPOL + pol) * DIMC + ch) * HWSZ + h * WW + w0 + pix];
    }
#pragma unroll
    for (int it = 0; it < 32; ++it) {
      int plane = it * 8 + hw;
      int pol = plane >> 6;
      int ch  = plane & 63;
      int mt = pix >> 2;
      int r  = ((pix & 3) << 2) | pol;
      int ks = ch >> 5;
      int hh;
      int ln = a_swz(r, ch & 31, &hh);
      ldsA[(((mt * 2 + ks) * 32 + ln) << 4) + hh] = (_Float16)xv[it];
    }
  }
  __syncthreads();
#endif

  // ---- Phase 2: q,k,v projections via v_wmma_f32_16x16x32_f16 --------------
  // All 12 independent tiles accumulate first (back-to-back WMMA issue, deep
  // ds-load pipelining), then a single convert/store pass — this removes the
  // WMMA->VALU RAW hazard NOPs and the per-tile s_wait_dscnt 0 stalls.
  {
    const int mt = wave;
    v16h a0 = *(const v16h*)(ldsA + (((mt * 2 + 0) * 32 + lane) << 4));
    v16h a1 = *(const v16h*)(ldsA + (((mt * 2 + 1) * 32 + lane) << 4));
    v8f acc[12];
#pragma unroll
    for (int m = 0; m < 3; ++m) {
#pragma unroll
      for (int nt = 0; nt < 4; ++nt) {
        v16h b0 = *(const v16h*)(ldsB + ((((m * 4 + nt) * 2 + 0) * 32 + lane) << 4));
        v16h b1 = *(const v16h*)(ldsB + ((((m * 4 + nt) * 2 + 1) * 32 + lane) << 4));
        v8f c = {};
        c = __builtin_amdgcn_wmma_f32_16x16x32_f16(false, a0, false, b0,
                                                   (short)0, c, false, false);
        c = __builtin_amdgcn_wmma_f32_16x16x32_f16(false, a1, false, b1,
                                                   (short)0, c, false, false);
        acc[m * 4 + nt] = c;
      }
    }
    // Single store pass. D layout: lane l, vgpr j -> (row = j+8*(l>=16), col=l%16)
#pragma unroll
    for (int m = 0; m < 3; ++m) {
#pragma unroll
      for (int nt = 0; nt < 4; ++nt) {
#pragma unroll
        for (int j = 0; j < 8; ++j) {
          int row = j + ((lane >> 4) << 3);
          int pix = mt * 4 + (row >> 2);
          int pol = row & 3;
          int ch  = nt * 16 + (lane & 15);
          ldsQKV[(((m * 16 + pix) * 4 + pol) << 6) + ch] = (_Float16)acc[m * 4 + nt][j];
        }
      }
    }
  }
  __syncthreads();

  // ---- Phase 3: per (pixel, head) attention in VALU ------------------------
  {
    const int pix  = tid >> 3;
    const int head = tid & 7;
    float qm[NPOL][8], km[NPOL][8], vm[NPOL][8];
#pragma unroll
    for (int p = 0; p < NPOL; ++p)
#pragma unroll
      for (int e = 0; e < 8; ++e) {
        int ch = head * 8 + e;
        qm[p][e] = (float)ldsQKV[(((0 * 16 + pix) * 4 + p) << 6) + ch];
        km[p][e] = (float)ldsQKV[(((1 * 16 + pix) * 4 + p) << 6) + ch];
        vm[p][e] = (float)ldsQKV[(((2 * 16 + pix) * 4 + p) << 6) + ch];
      }
#pragma unroll
    for (int p = 0; p < NPOL; ++p) {
      float nq = 0.f, nk = 0.f;
#pragma unroll
      for (int e = 0; e < 8; ++e) { nq += qm[p][e] * qm[p][e]; nk += km[p][e] * km[p][e]; }
      float iq = 1.0f / fmaxf(sqrtf(nq), EPSN);
      float ik = 1.0f / fmaxf(sqrtf(nk), EPSN);
#pragma unroll
      for (int e = 0; e < 8; ++e) { qm[p][e] *= iq; km[p][e] *= ik; }
    }
    float rs = rescale[head];
    float mbv[NPOL];
    size_t mbbase = (((size_t)b * HH + h) * WW + (w0 + pix)) * NPOL;
#pragma unroll
    for (int p = 0; p < NPOL; ++p) mbv[p] = mb[mbbase + p];

    float om[NPOL][8];
#pragma unroll
    for (int p = 0; p < NPOL; ++p) {
      float s[NPOL];
      float mx = -1e30f;
#pragma unroll
      for (int q = 0; q < NPOL; ++q) {
        float d = 0.f;
#pragma unroll
        for (int e = 0; e < 8; ++e) d += qm[p][e] * km[q][e];
        s[q] = d * rs + (mbv[q] - mbv[p]);
        mx = fmaxf(mx, s[q]);
      }
      float sum = 0.f;
#pragma unroll
      for (int q = 0; q < NPOL; ++q) { s[q] = __expf(s[q] - mx); sum += s[q]; }
      float inv = 1.0f / sum;
#pragma unroll
      for (int e = 0; e < 8; ++e) {
        float o = 0.f;
#pragma unroll
        for (int q = 0; q < NPOL; ++q) o += s[q] * vm[q][e];
        om[p][e] = o * inv;
      }
    }
    // attn-out back into A layout (reuse ldsA; all A reads are done)
    int mt = pix >> 2;
#pragma unroll
    for (int p = 0; p < NPOL; ++p) {
      int r = ((pix & 3) << 2) | p;
#pragma unroll
      for (int eg = 0; eg < 2; ++eg) {        // two packed groups of 4
        int ch0 = head * 8 + eg * 4;
        v4h h4;
#pragma unroll
        for (int e = 0; e < 4; ++e) h4[e] = (_Float16)om[p][eg * 4 + e];
        int ks = ch0 >> 5;
        int hh;
        int ln = a_swz(r, ch0 & 31, &hh);
        *(v4h*)(ldsA + (((mt * 2 + ks) * 32 + ln) << 4) + hh) = h4;
      }
    }
  }
  __syncthreads();

  // ---- Phase 4: output projection WMMA + bias (accumulate-then-store) ------
  {
    const int mt = wave;
    v16h a0 = *(const v16h*)(ldsA + (((mt * 2 + 0) * 32 + lane) << 4));
    v16h a1 = *(const v16h*)(ldsA + (((mt * 2 + 1) * 32 + lane) << 4));
    v8f acc[4];
#pragma unroll
    for (int nt = 0; nt < 4; ++nt) {
      v16h b0 = *(const v16h*)(ldsB + ((((3 * 4 + nt) * 2 + 0) * 32 + lane) << 4));
      v16h b1 = *(const v16h*)(ldsB + ((((3 * 4 + nt) * 2 + 1) * 32 + lane) << 4));
      v8f c = {};
      c = __builtin_amdgcn_wmma_f32_16x16x32_f16(false, a0, false, b0,
                                                 (short)0, c, false, false);
      c = __builtin_amdgcn_wmma_f32_16x16x32_f16(false, a1, false, b1,
                                                 (short)0, c, false, false);
      acc[nt] = c;
    }
#pragma unroll
    for (int nt = 0; nt < 4; ++nt) {
      float bias = b_proj[nt * 16 + (lane & 15)];   // col fixed per lane
#pragma unroll
      for (int j = 0; j < 8; ++j) {
        int row = j + ((lane >> 4) << 3);
        int pix = mt * 4 + (row >> 2);
        int pol = row & 3;
        int ch  = nt * 16 + (lane & 15);
        ldsOut[((pol * 64 + ch) << 4) + pix] = acc[nt][j] + bias;
      }
    }
  }
  __syncthreads();

  // ---- Phase 5: output store ----------------------------------------------
#if HAVE_ASYNC
  {
#pragma unroll
    for (int it = 0; it < 8; ++it) {
      int chunk = it * 128 + tid;       // 16B chunks, layout-identical LDS/global
      int plane = chunk >> 2;
      int quad  = chunk & 3;
      int pol = plane >> 6;
      int ch  = plane & 63;
      float* g = out + ((size_t)(b * NPOL + pol) * DIMC + ch) * HWSZ
                     + h * WW + w0 + quad * 4;
      async_l2g_b128(g, (const unsigned char*)ldsOut + chunk * 16);
    }
    WAIT_ASYNC();
  }
#else
  {
    const int hw  = tid >> 4;
    const int pix = tid & 15;
    for (int it = 0; it < 32; ++it) {
      int plane = it * 8 + hw;
      int pol = plane >> 6;
      int ch  = plane & 63;
      out[((size_t)(b * NPOL + pol) * DIMC + ch) * HWSZ + h * WW + w0 + pix] =
          ldsOut[(plane << 4) + pix];
    }
  }
#endif
}

// ---------------------------------------------------------------------------
extern "C" void kernel_launch(void* const* d_in, const int* in_sizes, int n_in,
                              void* d_out, int out_size, void* d_ws, size_t ws_size,
                              hipStream_t stream) {
  (void)in_sizes; (void)n_in; (void)out_size; (void)ws_size;
  const float* x_in    = (const float*)d_in[0];
  const float* mask    = (const float*)d_in[1];
  const float* wq      = (const float*)d_in[2];
  const float* wk      = (const float*)d_in[3];
  const float* wv      = (const float*)d_in[4];
  const float* w_proj  = (const float*)d_in[5];
  const float* b_proj  = (const float*)d_in[6];
  const float* rescale = (const float*)d_in[7];
  const float* dw_w    = (const float*)d_in[8];
  const float* dw_b    = (const float*)d_in[9];
  const float* pw_w    = (const float*)d_in[10];
  const float* pw_b    = (const float*)d_in[11];
  float* out = (float*)d_out;

  float* ws    = (float*)d_ws;   // [0..64) w_avg, [64] b_avg
  float* mbbuf = ws + 256;       // [4][128][128][4] floats = 1 MB

  prep_kernel<<<1, 128, 0, stream>>>(pw_w, pw_b, ws);
  maskbias_kernel<<<(NB * NPOL * HWSZ) / 256, 256, 0, stream>>>(mask, dw_w, dw_b, ws, mbbuf);
  attention_kernel<<<NB * HH * (WW / 16), 128, 0, stream>>>(
      x_in, wq, wk, wv, w_proj, b_proj, rescale, mbbuf, out);
}
